// TransformerBlock_3530463117491
// MI455X (gfx1250) — matrix operand order repaired and checked by reading, not compiled, not run on hardware
//
#include <hip/hip_runtime.h>
#include <stdint.h>

typedef unsigned short u16;
typedef __attribute__((ext_vector_type(16))) __bf16          v16bf;
typedef __attribute__((ext_vector_type(16))) unsigned short  v16u;
typedef __attribute__((ext_vector_type(8)))  float           v8f;
typedef __attribute__((ext_vector_type(4)))  unsigned        v4u;
typedef __attribute__((ext_vector_type(8)))  int             v8i;
typedef __attribute__((ext_vector_type(4)))  int             v4i;

#define D_   768
#define H_   12
#define DH_  64
#define DFF_ 3072
#define B_   2
#define S_   2048
#define MTOK (B_ * S_)   // 4096 tokens

#if __has_builtin(__builtin_amdgcn_tensor_load_to_lds)
#define HAVE_TDM 1
#else
#define HAVE_TDM 0
#endif
#if __has_include(<hip/amd_detail/amd_gfx1250_TDM.h>)
#define TDM_6ARGS 1   // amdgpu-toolchain / therock-10.0 headers -> 6-arg builtin
#else
#define TDM_6ARGS 0   // ROCm 7.2 -> 5-arg builtin
#endif

// ---------- helpers ----------

__device__ inline u16 f2bf(float f) {
    return __builtin_bit_cast(u16, static_cast<__bf16>(f));
}

// pack two floats into one dword of bf16 pair (lowers to v_cvt_pk_bf16_f32)
__device__ inline unsigned f2bf_pk(float lo, float hi) {
    return (unsigned)f2bf(lo) | ((unsigned)f2bf(hi) << 16);
}

__device__ inline v8f v8f_zero() {
    v8f z;
#pragma unroll
    for (int i = 0; i < 8; ++i) z[i] = 0.0f;
    return z;
}

__device__ inline v8f wmma_bf16(v16u a, v16u b, v8f c) {
    return __builtin_amdgcn_wmma_f32_16x16x32_bf16(
        false, __builtin_bit_cast(v16bf, a),
        false, __builtin_bit_cast(v16bf, b),
        (short)0, c, false, false);
}

struct U32x8 { uint4 a, b; };   // 32 bytes == one v16u fragment

// A-fragment (16-bit 16x32, ISA 7.12.2): per lane two contiguous 8-elem runs:
//   elems 0..7  -> K = 8*half + 0..7
//   elems 8..15 -> K = 16 + 8*half + 0..7
__device__ inline v16u afrag_ld(const u16* tile, int ld, int lane) {
    const int m = lane & 15, half = lane >> 4;
    const u16* base = tile + m * ld + 8 * half;
    U32x8 t;
    t.a = *(const uint4*)(base);
    t.b = *(const uint4*)(base + 16);
    return __builtin_bit_cast(v16u, t);
}

// B-fragment from TRANSPOSED LDS tile Bt[n][k] (ld = 32): elem e<8 -> K = 8*half + e, e>=8 -> K = 16 + 8*half + (e-8)
__device__ inline v16u bfrag_t_ld(const u16* Bt, int col0, int lane) {
    const int n = lane & 15, half = lane >> 4;
    const u16* base = Bt + (col0 + n) * 32 + 8 * half;
    U32x8 t;
    t.a = *(const uint4*)(base);
    t.b = *(const uint4*)(base + 16);
    return __builtin_bit_cast(v16u, t);
}

// B-fragment from row-major K tile Ks[key][dh] for Q*K^T (contiguous over dh)
__device__ inline v16u bfrag_ks(const u16* Ks, int sub, int f, int lane) {
    const int n = lane & 15, half = lane >> 4;
    const u16* base = Ks + (sub * 16 + n) * 64 + f * 32 + 8 * half;
    U32x8 t;
    t.a = *(const uint4*)(base);
    t.b = *(const uint4*)(base + 16);
    return __builtin_bit_cast(v16u, t);
}

// ---------- Tensor Data Mover: 2D tile (2-byte elements) global -> LDS ----------
// D# per CDNA5 ISA. Issued by one wave; EXEC ignored.

__device__ inline void tdm_load_tile_2d_b16(unsigned lds_off, const void* gptr,
                                            unsigned tile_d0, unsigned tile_d1,
                                            unsigned tensor_d0, unsigned tensor_d1,
                                            unsigned long long stride_d0) {
#if HAVE_TDM
    unsigned long long ga = (unsigned long long)gptr;
    v4u g0;
    g0[0] = 1u;                                    // count=1, user descriptor
    g0[1] = lds_off;                               // lds_addr
    g0[2] = (unsigned)ga;                          // global_addr[31:0]
    g0[3] = ((unsigned)(ga >> 32) & 0x01FFFFFFu)   // global_addr[56:32]
            | 0x80000000u;                         // type=2 (image)
    v8i g1;
    g1[0] = (int)(1u << 16);                                        // data_size=1 (2B)
    g1[1] = (int)((tensor_d0 & 0xFFFFu) << 16);                     // tensor_dim0 lo
    g1[2] = (int)((tensor_d0 >> 16) | ((tensor_d1 & 0xFFFFu) << 16));
    g1[3] = (int)((tensor_d1 >> 16) | (tile_d0 << 16));             // tile_dim0
    g1[4] = (int)tile_d1;                                           // tile_dim1 (tile_dim2=0)
    g1[5] = (int)(unsigned)(stride_d0 & 0xFFFFFFFFull);             // dim0 stride lo
    g1[6] = (int)(unsigned)(stride_d0 >> 32);                       // dim0 stride hi
    g1[7] = 0;
    v4i z = {0, 0, 0, 0};
#if TDM_6ARGS
    v8i z8 = {0, 0, 0, 0, 0, 0, 0, 0};
    __builtin_amdgcn_tensor_load_to_lds(g0, g1, z, z, z8, 0);
#else
    __builtin_amdgcn_tensor_load_to_lds(g0, g1, z, z, 0);
#endif
#else
    (void)lds_off; (void)gptr; (void)tile_d0; (void)tile_d1;
    (void)tensor_d0; (void)tensor_d1; (void)stride_d0;
#endif
}

__device__ inline void tdm_wait() {
#if HAVE_TDM
    __builtin_amdgcn_s_wait_tensorcnt((short)0);
#endif
}

__device__ inline unsigned lds_off_of(const void* p) {
    return (unsigned)(unsigned long long)p;   // LDS aperture: addr[31:0] is the LDS offset
}

// ---------- LayerNorm -> bf16 ----------

__global__ void __launch_bounds__(256)
ln_kernel(const float* __restrict__ x, const float* __restrict__ gain,
          const float* __restrict__ off, u16* __restrict__ out) {
    __shared__ float redS[8], redQ[8];
    const int row = blockIdx.x;
    const int tid = threadIdx.x;
    const float* xr = x + (size_t)row * D_;
    float v[3], s = 0.0f, q = 0.0f;
#pragma unroll
    for (int j = 0; j < 3; ++j) {
        v[j] = xr[tid + 256 * j];
        s += v[j];
        q += v[j] * v[j];
    }
#pragma unroll
    for (int mm = 16; mm >= 1; mm >>= 1) {
        s += __shfl_xor(s, mm);
        q += __shfl_xor(q, mm);
    }
    if ((tid & 31) == 0) { redS[tid >> 5] = s; redQ[tid >> 5] = q; }
    __syncthreads();
    s = 0.0f; q = 0.0f;
#pragma unroll
    for (int j = 0; j < 8; ++j) { s += redS[j]; q += redQ[j]; }
    const float mean = s * (1.0f / D_);
    float var = fmaxf(q * (1.0f / D_) - mean * mean, 0.0f);
    const float inv = 1.0f / (sqrtf(var) + 1e-5f);
#pragma unroll
    for (int j = 0; j < 3; ++j) {
        int c = tid + 256 * j;
        out[(size_t)row * D_ + c] = f2bf(gain[c] * ((v[j] - mean) * inv) + off[c]);
    }
}

// ---------- 128x64 WMMA GEMM tile core (256 thr = 8 waves, each wave 32x32) ----------
// A: bf16 [M x K] row-major. B: fp32 [K x ldb] row-major, col offset n0.
// A tile loaded by TDM (wave 0) when available; B tile converted + TRANSPOSED into LDS
// (each thread: 2 k-rows x 4 cols -> packed bf16 pairs, ds_store_b32).

__device__ inline void gemm_tile_128x64(const u16* __restrict__ A, int lda, int m0,
                                        const float* __restrict__ Bw, int ldb, int n0,
                                        int K, u16* As, u16* Bt, v8f c[2][2]) {
    const int tid  = threadIdx.x;
    const int lane = tid & 31;
    const int w    = tid >> 5;
    const int wm   = w & 3;    // wave row 0..3 -> m offset 32*wm
    const int wn   = w >> 2;   // wave col 0..1 -> n offset 32*wn

    c[0][0] = v8f_zero(); c[0][1] = v8f_zero();
    c[1][0] = v8f_zero(); c[1][1] = v8f_zero();

    const int ar  = tid >> 1, ac = (tid & 1) * 16;  // A fallback copy: 16 elems/thread
    const int bk0 = (tid & 15) * 2;                 // B: k-row pair 0,2,..,30
    const int bn0 = (tid >> 4) * 4;                 // B: 4 cols 0,4,..,60

    const unsigned as_off = lds_off_of(As);
    const int nk = K / 32;
    for (int kt = 0; kt < nk; ++kt) {
#if HAVE_TDM
        if (w == 0)   // async A tile: 128 rows x 32 cols of bf16, row stride lda
            tdm_load_tile_2d_b16(as_off, A + (size_t)m0 * lda + kt * 32,
                                 32, 128, (unsigned)K, (unsigned)MTOK,
                                 (unsigned long long)lda);
#else
        {
            const uint4* src = (const uint4*)(A + (size_t)(m0 + ar) * lda + kt * 32 + ac);
            uint4* dst = (uint4*)(As + ar * 32 + ac);
            dst[0] = src[0];
            dst[1] = src[1];
        }
#endif
        {   // B tile 32x64 fp32 -> bf16, transposed Bt[n][k]; pack k-pairs into b32
            const float* sp = Bw + (size_t)(kt * 32 + bk0) * ldb + n0 + bn0;
            float4 r0 = *(const float4*)(sp);
            float4 r1 = *(const float4*)(sp + ldb);
            unsigned* bt32 = (unsigned*)(Bt + bn0 * 32 + bk0);  // stride 16 dwords per n
            bt32[0]  = f2bf_pk(r0.x, r1.x);
            bt32[16] = f2bf_pk(r0.y, r1.y);
            bt32[32] = f2bf_pk(r0.z, r1.z);
            bt32[48] = f2bf_pk(r0.w, r1.w);
        }
        if (kt + 1 < nk)
            __builtin_prefetch(Bw + (size_t)((kt + 1) * 32 + bk0) * ldb + n0 + bn0, 0, 0);
#if HAVE_TDM
        if (w == 0) tdm_wait();
#endif
        __syncthreads();

#pragma unroll
        for (int mi = 0; mi < 2; ++mi) {
            v16u a = afrag_ld(As + (wm * 32 + mi * 16) * 32, 32, lane);
#pragma unroll
            for (int ni = 0; ni < 2; ++ni) {
                v16u b = bfrag_t_ld(Bt, wn * 32 + ni * 16, lane);
                c[mi][ni] = wmma_bf16(a, b, c[mi][ni]);
            }
        }
        __syncthreads();
    }
}

// ---------- generic GEMM + bias (+relu) (+residual) ----------

template <bool RELU, bool RESID, bool OUT_BF16>
__global__ void __launch_bounds__(256)
gemm_bias_kernel(const u16* __restrict__ A, const float* __restrict__ Bw,
                 const float* __restrict__ bias, const float* __restrict__ resid,
                 void* __restrict__ Out, int M, int N, int K) {
    __shared__ u16 As[128 * 32];
    __shared__ u16 Bt[64 * 32];
    const int m0 = blockIdx.x * 128;
    const int n0 = blockIdx.y * 64;
    v8f c[2][2];
    gemm_tile_128x64(A, K, m0, Bw, N, n0, K, As, Bt, c);

    const int lane = threadIdx.x & 31;
    const int w = threadIdx.x >> 5;
    const int wm = w & 3, wn = w >> 2;
    const int nn = lane & 15, half = lane >> 4;
#pragma unroll
    for (int mi = 0; mi < 2; ++mi)
#pragma unroll
        for (int ni = 0; ni < 2; ++ni)
#pragma unroll
            for (int i = 0; i < 8; ++i) {
                int mg = m0 + wm * 32 + mi * 16 + i + 8 * half;
                int ng = n0 + wn * 32 + ni * 16 + nn;
                float v = c[mi][ni][i] + bias[ng];
                if (RELU)  v = fmaxf(v, 0.0f);
                if (RESID) v += resid[(size_t)mg * N + ng];
                if (OUT_BF16) ((u16*)Out)[(size_t)mg * N + ng] = f2bf(v);
                else          ((float*)Out)[(size_t)mg * N + ng] = v;
            }
}

// ---------- QKV projection (per head, per q/k/v via blockIdx.z) ----------

__global__ void __launch_bounds__(256)
qkv_kernel(const u16* __restrict__ A,
           const float* __restrict__ Wq, const float* __restrict__ bq,
           const float* __restrict__ Wk, const float* __restrict__ bk,
           const float* __restrict__ Wv, const float* __restrict__ bv,
           u16* __restrict__ Qo, u16* __restrict__ Ko, u16* __restrict__ Vo) {
    __shared__ u16 As[128 * 32];
    __shared__ u16 Bt[64 * 32];
    const int h = blockIdx.y;
    const int which = blockIdx.z;
    const float* W  = (which == 0) ? Wq : (which == 1) ? Wk : Wv;
    const float* bb = (which == 0) ? bq : (which == 1) ? bk : bv;
    u16* Out        = (which == 0) ? Qo : (which == 1) ? Ko : Vo;
    W  += (size_t)h * D_ * DH_;
    bb += h * DH_;

    const int m0 = blockIdx.x * 128;
    v8f c[2][2];
    gemm_tile_128x64(A, D_, m0, W, DH_, 0, D_, As, Bt, c);

    const int lane = threadIdx.x & 31;
    const int w = threadIdx.x >> 5;
    const int wm = w & 3, wn = w >> 2;
    const int nn = lane & 15, half = lane >> 4;
#pragma unroll
    for (int mi = 0; mi < 2; ++mi)
#pragma unroll
        for (int ni = 0; ni < 2; ++ni)
#pragma unroll
            for (int i = 0; i < 8; ++i) {
                int mg = m0 + wm * 32 + mi * 16 + i + 8 * half;   // token 0..4095
                int ng = wn * 32 + ni * 16 + nn;                  // dh 0..63
                int bi = mg >> 11;                                // /2048
                int s  = mg & 2047;
                float v = c[mi][ni][i] + bb[ng];
                Out[(size_t)((bi * H_ + h) * S_ + s) * DH_ + ng] = f2bf(v);
            }
}

// ---------- causal flash attention (one wave = 16 query rows) ----------

__global__ void __launch_bounds__(256)
attn_kernel(const u16* __restrict__ Q, const u16* __restrict__ K,
            const u16* __restrict__ V, u16* __restrict__ O) {
    __shared__ u16 Ks[32 * 64];      // key-major (scores B-frag contiguous over dh)
    __shared__ u16 Vt[64 * 32];      // dh-major  (P.V  B-frag contiguous over key)
    __shared__ u16 Ps[8][16 * 32];   // per-wave P tile (16 queries x 32 keys)

    const int tid  = threadIdx.x;
    const int lane = tid & 31;
    const int w    = tid >> 5;
    const int nn   = lane & 15;
    const int half = lane >> 4;

    const int bh = blockIdx.y;        // b*H + h
    const int b  = bh / H_;
    const int h  = bh % H_;
    const int q0blk = blockIdx.x * 128;
    const int qbw   = q0blk + 16 * w; // this wave's query base

    const u16* Qg = Q + (size_t)bh * S_ * DH_;
    const u16* Kg = K + (size_t)bh * S_ * DH_;
    const u16* Vg = V + (size_t)bh * S_ * DH_;

    // Q fragments: 16 rows x 64 dh as two 16x32 A-frags (vector loads, loaded once)
    v16u aq[2];
#pragma unroll
    for (int f = 0; f < 2; ++f) {
        const u16* base = Qg + (size_t)(qbw + nn) * DH_ + f * 32 + 8 * half;
        U32x8 t;
        t.a = *(const uint4*)(base);
        t.b = *(const uint4*)(base + 16);
        aq[f] = __builtin_bit_cast(v16u, t);
    }

    v8f o[4];
#pragma unroll
    for (int t = 0; t < 4; ++t) o[t] = v8f_zero();
    float m_run[8], l_run[8];
#pragma unroll
    for (int i = 0; i < 8; ++i) { m_run[i] = -1e30f; l_run[i] = 0.0f; }

    const int vk0 = (tid >> 4) * 2;   // V transpose: key pair 0,2,..,30
    const int vd0 = (tid & 15) * 4;   // V transpose: 4 dh 0,4,..,60

    const unsigned ks_off = lds_off_of(Ks);
    const int nkt = q0blk / 32 + 4;   // key tiles of 32, up to causal limit of block
    for (int kt = 0; kt < nkt; ++kt) {
        const int kb = kt * 32;
#if HAVE_TDM
        if (w == 0)   // K tile 32 keys x 64 dh, contiguous rows -> TDM async copy
            tdm_load_tile_2d_b16(ks_off, Kg + (size_t)kb * DH_,
                                 64, 32, 64, (unsigned)S_, 64ull);
#else
        ((uint4*)Ks)[tid] = ((const uint4*)(Kg + (size_t)kb * DH_))[tid];
#endif
        {   // V tile, stored transposed Vt[dh][key]; pack key-pairs into b32 stores
            const u16* sp = Vg + (size_t)(kb + vk0) * DH_ + vd0;
            uint2 r0 = *(const uint2*)(sp);          // key vk0:   dh vd0..vd0+3
            uint2 r1 = *(const uint2*)(sp + DH_);    // key vk0+1: dh vd0..vd0+3
            unsigned* vt32 = (unsigned*)(Vt + vd0 * 32 + vk0);   // stride 16 dwords per dh
            vt32[0]  = (r0.x & 0xFFFFu) | (r1.x << 16);
            vt32[16] = (r0.x >> 16) | (r1.x & 0xFFFF0000u);
            vt32[32] = (r0.y & 0xFFFFu) | (r1.y << 16);
            vt32[48] = (r0.y >> 16) | (r1.y & 0xFFFF0000u);
        }
#if HAVE_TDM
        if (w == 0) tdm_wait();
#endif
        __syncthreads();

        if (kb <= qbw + 15) {   // wave-uniform: any unmasked work in this tile?
            // scores: two 16x16 fragments over 32 keys, K-dim = dh (2 chunks of 32)
            v8f s0 = v8f_zero(), s1 = v8f_zero();
#pragma unroll
            for (int f = 0; f < 2; ++f) {
                s0 = wmma_bf16(aq[f], bfrag_ks(Ks, 0, f, lane), s0);
                s1 = wmma_bf16(aq[f], bfrag_ks(Ks, 1, f, lane), s1);
            }

            float p0[8], p1[8], alpha[8];
#pragma unroll
            for (int i = 0; i < 8; ++i) {
                const int qrow = qbw + i + 8 * half;
                float v0 = s0[i] * 0.125f;        // 1/sqrt(64)
                float v1 = s1[i] * 0.125f;
                const bool ok0 = (kb + nn)      <= qrow;
                const bool ok1 = (kb + 16 + nn) <= qrow;
                v0 = ok0 ? v0 : -1e30f;
                v1 = ok1 ? v1 : -1e30f;
                float rm = fmaxf(v0, v1);
#pragma unroll
                for (int mm = 8; mm >= 1; mm >>= 1) rm = fmaxf(rm, __shfl_xor(rm, mm));
                const float mnew = fmaxf(m_run[i], rm);
                alpha[i] = __expf(m_run[i] - mnew);
                p0[i] = ok0 ? __expf(v0 - mnew) : 0.0f;
                p1[i] = ok1 ? __expf(v1 - mnew) : 0.0f;
                float rs = p0[i] + p1[i];
#pragma unroll
                for (int mm = 8; mm >= 1; mm >>= 1) rs += __shfl_xor(rs, mm);
                l_run[i] = l_run[i] * alpha[i] + rs;
                m_run[i] = mnew;
            }
#pragma unroll
            for (int t = 0; t < 4; ++t)
#pragma unroll
                for (int i = 0; i < 8; ++i) o[t][i] *= alpha[i];

            // write P (C-layout) to per-wave LDS (pack subtile pairs), reload as A-frag
            u16* Pw = Ps[w];
#pragma unroll
            for (int i = 0; i < 8; ++i) {
                Pw[(i + 8 * half) * 32 + nn]      = f2bf(p0[i]);
                Pw[(i + 8 * half) * 32 + 16 + nn] = f2bf(p1[i]);
            }
            asm volatile("s_wait_dscnt 0" ::: "memory");

            v16u ap = afrag_ld(Pw, 32, lane);
#pragma unroll
            for (int t = 0; t < 4; ++t)
                o[t] = wmma_bf16(ap, bfrag_t_ld(Vt, t * 16, lane), o[t]);
        }
        __syncthreads();
    }

    // normalize and write concatenated-head output [token, h*64+dh] as bf16
#pragma unroll
    for (int t = 0; t < 4; ++t)
#pragma unroll
        for (int i = 0; i < 8; ++i) {
            const int q = qbw + i + 8 * half;
            const float val = o[t][i] / l_run[i];
            O[(size_t)(b * S_ + q) * D_ + h * DH_ + t * 16 + nn] = f2bf(val);
        }
}

// ---------- host launcher ----------

extern "C" void kernel_launch(void* const* d_in, const int* in_sizes, int n_in,
                              void* d_out, int out_size, void* d_ws, size_t ws_size,
                              hipStream_t stream) {
    (void)in_sizes; (void)n_in; (void)out_size; (void)ws_size;
    const float* X  = (const float*)d_in[0];
    const float* Wq = (const float*)d_in[1];
    const float* bq = (const float*)d_in[2];
    const float* Wk = (const float*)d_in[3];
    const float* bk = (const float*)d_in[4];
    const float* Wv = (const float*)d_in[5];
    const float* bv = (const float*)d_in[6];
    const float* W0 = (const float*)d_in[7];
    const float* b0 = (const float*)d_in[8];
    const float* W1 = (const float*)d_in[9];
    const float* b1 = (const float*)d_in[10];
    const float* W2 = (const float*)d_in[11];
    const float* b2 = (const float*)d_in[12];
    const float* g1 = (const float*)d_in[13];
    const float* o1 = (const float*)d_in[14];
    const float* g2 = (const float*)d_in[15];
    const float* o2 = (const float*)d_in[16];

    char* ws = (char*)d_ws;
    size_t off = 0;
    auto alloc = [&](size_t bytes) -> char* {
        char* p = ws + off;
        off += (bytes + 255) & ~(size_t)255;
        return p;
    };
    u16*   ln1  = (u16*)alloc((size_t)MTOK * D_ * 2);          // bf16 LN1(X)
    u16*   qb   = (u16*)alloc((size_t)B_ * H_ * S_ * DH_ * 2); // bf16 q
    u16*   kb   = (u16*)alloc((size_t)B_ * H_ * S_ * DH_ * 2); // bf16 k
    u16*   vb   = (u16*)alloc((size_t)B_ * H_ * S_ * DH_ * 2); // bf16 v
    u16*   att  = (u16*)alloc((size_t)MTOK * D_ * 2);          // bf16 attn out (heads concat)
    float* hbuf = (float*)alloc((size_t)MTOK * D_ * 4);        // fp32 residual h
    u16*   y2   = (u16*)alloc((size_t)MTOK * D_ * 2);          // bf16 LN2(h)
    u16*   ff1  = (u16*)alloc((size_t)MTOK * DFF_ * 2);        // bf16 relu(FFN1)

    // 1) LN1
    ln_kernel<<<MTOK, 256, 0, stream>>>(X, g1, o1, ln1);
    // 2) QKV projections (grid.z selects q/k/v)
    qkv_kernel<<<dim3(MTOK / 128, H_, 3), 256, 0, stream>>>(
        ln1, Wq, bq, Wk, bk, Wv, bv, qb, kb, vb);
    // 3) causal flash attention
    attn_kernel<<<dim3(S_ / 128, B_ * H_), 256, 0, stream>>>(qb, kb, vb, att);
    // 4) output projection + residual: h = X + att @ W0 + b0  (fp32)
    gemm_bias_kernel<false, true, false><<<dim3(MTOK / 128, D_ / 64), 256, 0, stream>>>(
        att, W0, b0, X, hbuf, MTOK, D_, D_);
    // 5) LN2
    ln_kernel<<<MTOK, 256, 0, stream>>>(hbuf, g2, o2, y2);
    // 6) FFN1 + ReLU (bf16 out)
    gemm_bias_kernel<true, false, true><<<dim3(MTOK / 128, DFF_ / 64), 256, 0, stream>>>(
        y2, W1, b1, nullptr, ff1, MTOK, DFF_, D_);
    // 7) FFN2 + residual -> d_out (fp32)
    gemm_bias_kernel<false, true, false><<<dim3(MTOK / 128, D_ / 64), 256, 0, stream>>>(
        ff1, W2, b2, hbuf, (float*)d_out, MTOK, D_, DFF_);
}
